// CostVolume_7164005450223
// MI455X (gfx1250) — matrix operand order repaired and checked by reading, not compiled, not run on hardware
//
#include <hip/hip_runtime.h>
#include <hip/hip_bf16.h>

typedef _Float16 v8h  __attribute__((ext_vector_type(8)));
typedef _Float16 v16h __attribute__((ext_vector_type(16)));
typedef float    v8f  __attribute__((ext_vector_type(8)));

#define CAT16(lo, hi) __builtin_shufflevector((lo), (hi), 0,1,2,3,4,5,6,7,8,9,10,11,12,13,14,15)

// Problem constants (from reference)
constexpr int B_ = 8, C_ = 256, H_ = 96, W_ = 192;
constexpr int D_ = 9;              // 2*DELTA+1
constexpr float EPS_ = 1e-8f;

// Tiling
constexpr int WTILE  = 16;         // pixels per WMMA row-tile
constexpr int NWT    = W_ / WTILE; // 12 w-tiles
constexpr int HCHUNK = 24;         // rows per workgroup
constexpr int NHC    = H_ / HCHUNK;// 4 h-chunks
constexpr int NPOS   = 28;         // x2 positions staged: w0-4 .. w0+23
constexpr int CHS    = 264;        // padded channel stride in halves (264*2B = 528B, 16B aligned, bank-spread)

__global__ __launch_bounds__(128)
void cost_volume_wmma(const float* __restrict__ img1,
                      const float* __restrict__ img2,
                      float* __restrict__ out) {
    __shared__ _Float16 x1b[WTILE * CHS];        // 8448 B   img1 row tile (f16)
    __shared__ _Float16 x2b[9 * NPOS * CHS];     // 133056 B rolling 9-row img2 window (f16)
    __shared__ float    n1s[WTILE];              // norms of img1 pixels (fp32)
    __shared__ float    n2s[9 * NPOS];           // norms of img2 positions (fp32)
    __shared__ float    ss1[128];                // sum-of-squares partials
    __shared__ float    ss2[128];

    const int tid  = threadIdx.x;   // 0..127 (4 waves)
    const int lane = tid & 31;
    const int wv   = tid >> 5;

    const int bidx = blockIdx.x;
    const int hc = bidx & 3;                 // NHC = 4
    const int wt = (bidx >> 2) % NWT;
    const int b  = bidx / (4 * NWT);
    const int h0 = hc * HCHUNK;
    const int w0 = wt * WTILE;

    // ---- loader for one img2 row (zero-filled outside the image) ----
    auto loadX2 = [&](int r, int s) {
        const int i  = tid & 31;    // position index 0..31 (use < 28)
        const int cb = tid >> 5;    // 0..3
        float ss = 0.f;
        if (i < NPOS) {
            const int w = w0 - 4 + i;
            const bool inb = (r >= 0) && (r < H_) && (w >= 0) && (w < W_);
            _Float16* dst = &x2b[(s * NPOS + i) * CHS];
            #pragma unroll 4
            for (int k = 0; k < 64; ++k) {
                const int c = cb + (k << 2);
                float x = 0.f;
                if (inb) x = img2[(((size_t)b * C_ + c) * H_ + r) * W_ + w];
                ss += x * x;
                dst[c] = (_Float16)x;
            }
        }
        ss2[(cb << 5) + i] = ss;
    };

    // ---- prime the rolling window with rows h0-4 .. h0+4 ----
    for (int rr = 0; rr < 9; ++rr) {
        const int r = h0 - 4 + rr;
        const int s = (r + 9) % 9;
        loadX2(r, s);
        __syncthreads();
        if (tid < NPOS) {
            const float t = ss2[tid] + ss2[32 + tid] + ss2[64 + tid] + ss2[96 + tid];
            n2s[s * NPOS + tid] = fmaxf(sqrtf(t), EPS_);
        }
        __syncthreads();
    }

    // ---- march down the h-chunk ----
    for (int h = h0; h < h0 + HCHUNK; ++h) {
        // load img1 row tile (16 pixels x 256 ch), fp32->f16, fp32 sum-of-squares
        {
            const int i  = tid & 15;
            const int cb = tid >> 4;
            float ss = 0.f;
            _Float16* dst = &x1b[i * CHS];
            #pragma unroll 4
            for (int k = 0; k < 32; ++k) {
                const int c = cb + (k << 3);
                const float x = img1[(((size_t)b * C_ + c) * H_ + h) * W_ + (w0 + i)];
                ss += x * x;
                dst[c] = (_Float16)x;
            }
            ss1[(cb << 4) + i] = ss;
        }
        // roll the window: bring in img2 row h+4 (zeros past the bottom)
        if (h > h0) loadX2(h + 4, (h + 13) % 9);
        __syncthreads();
        if (tid < WTILE) {
            float t = 0.f;
            #pragma unroll
            for (int j = 0; j < 8; ++j) t += ss1[(j << 4) + tid];
            n1s[tid] = fmaxf(sqrtf(t), EPS_);
        } else if (h > h0 && tid >= 32 && tid < 32 + NPOS) {
            const int p = tid - 32;
            const int s = (h + 13) % 9;
            const float t = ss2[p] + ss2[32 + p] + ss2[64 + p] + ss2[96 + p];
            n2s[s * NPOS + p] = fmaxf(sqrtf(t), EPS_);
        }
        __syncthreads();

        // ---- WMMA compute: 4 waves split the 9 dh rows ----
        const int m   = lane & 15;   // A row / B col / D col index
        const int sel = lane >> 4;
        for (int dh = wv; dh < D_; dh += 4) {
            const int r    = h + dh - 4;
            const int slot = (r + 9) % 9;
            v8f acc1 = {0.f, 0.f, 0.f, 0.f, 0.f, 0.f, 0.f, 0.f};
            v8f acc2 = {0.f, 0.f, 0.f, 0.f, 0.f, 0.f, 0.f, 0.f};
            #pragma unroll
            for (int kb = 0; kb < 8; ++kb) {
                // A fragment: lane<16 holds K 0..15; lane>=16 holds K 16..31
                const int ka = kb * 32 + (sel << 4);
                const v8h a0 = *(const v8h*)&x1b[m * CHS + ka];
                const v8h a1 = *(const v8h*)&x1b[m * CHS + ka + 8];
                const v16h A = CAT16(a0, a1);
                // B fragments: lane<16 holds K 0..15; lane>=16 holds K 16..31 of column m
                const int kbse = kb * 32 + (sel << 4);
                const _Float16* b1p = &x2b[(slot * NPOS + m) * CHS + kbse];
                const _Float16* b2p = &x2b[(slot * NPOS + 12 + m) * CHS + kbse];
                const v16h B1 = CAT16(*(const v8h*)b1p, *(const v8h*)(b1p + 8));
                const v16h B2 = CAT16(*(const v8h*)b2p, *(const v8h*)(b2p + 8));
                acc1 = __builtin_amdgcn_wmma_f32_16x16x32_f16(false, A, false, B1,
                                                              (short)0, acc1, false, false);
                acc2 = __builtin_amdgcn_wmma_f32_16x16x32_f16(false, A, false, B2,
                                                              (short)0, acc2, false, false);
            }
            // ---- banded output stores with normalization ----
            // D layout: VGPR v, lanes 0-15 -> M=v, lanes 16-31 -> M=v+8; N = lane&15
            const int j = m;
            #pragma unroll
            for (int v = 0; v < 8; ++v) {
                const int i = v + (sel << 3);            // pixel index in tile
                const int dw1 = j - i;                   // tile1: col j = img2 pos w0-4+j
                if (dw1 >= 0 && dw1 <= 8 && j <= 11) {
                    const float denom = n1s[i] * n2s[slot * NPOS + j];
                    const size_t o = ((((size_t)b * H_ + h) * W_ + (w0 + i)) * D_ + dh) * D_ + dw1;
                    out[o] = acc1[v] / denom;
                }
                const int dw2 = j - i + 12;              // tile2: col j = img2 pos w0+8+j
                if (dw2 >= 0 && dw2 <= 8) {
                    const float denom = n1s[i] * n2s[slot * NPOS + 12 + j];
                    const size_t o = ((((size_t)b * H_ + h) * W_ + (w0 + i)) * D_ + dh) * D_ + dw2;
                    out[o] = acc2[v] / denom;
                }
            }
        }
        __syncthreads();   // before the next h overwrites x1b / a window slot
    }
}

extern "C" void kernel_launch(void* const* d_in, const int* in_sizes, int n_in,
                              void* d_out, int out_size, void* d_ws, size_t ws_size,
                              hipStream_t stream) {
    (void)in_sizes; (void)n_in; (void)d_ws; (void)ws_size; (void)out_size;
    const float* img1 = (const float*)d_in[0];
    const float* img2 = (const float*)d_in[1];
    float* out = (float*)d_out;
    const int grid = B_ * NWT * NHC;   // 8 * 12 * 4 = 384 workgroups
    cost_volume_wmma<<<grid, 128, 0, stream>>>(img1, img2, out);
}
